// TTmerNet_65077344469254
// MI455X (gfx1250) — matrix operand order repaired and checked by reading, not compiled, not run on hardware
//
#include <hip/hip_runtime.h>
#include <hip/hip_bf16.h>
#include <math.h>

typedef __attribute__((ext_vector_type(16))) _Float16 v16h;
typedef __attribute__((ext_vector_type(8)))  float    v8f;

#define NN    200000
#define EE    1600000
#define CC    128
#define NTTs  20000
#define NMOLs 2000

// ---------------------------------------------------------------------------
// Weight packing: W is [OC, 128] row-major (torch Linear). We compute
// t = x @ W^T, i.e. B[k][n] = W[n][k]. Pack into WMMA 16-bit B-fragment
// layout: lane L holds column n = L&15, k = kt*32 + 16*(L>>4) + i (i=0..15).
// Storage: Bp[((nt*4 + kt)*32 + lane)*16 + i]
// ---------------------------------------------------------------------------
__global__ void ttm_pack_w(const float* __restrict__ W, _Float16* __restrict__ Bp) {
    int nt = blockIdx.x;            // n-tile
    int lane = threadIdx.x;         // 0..31
    int n = nt * 16 + (lane & 15);
    int khalf = lane >> 4;
    for (int kt = 0; kt < 4; ++kt) {
        _Float16* dst = Bp + (((nt * 4 + kt) * 32 + lane) * 16);
        #pragma unroll
        for (int i = 0; i < 16; ++i) {
            int k = kt * 32 + khalf * 16 + i;
            dst[i] = (_Float16)W[n * 128 + k];
        }
    }
}

// ---------------------------------------------------------------------------
// WMMA GEMM, templated: Out[M, NT*16] = X[M,128] @ Bpacked + bias (opt relu).
// Each wave owns TWO 16-row strips (32 rows) with independent accumulator
// chains sharing the same B fragments: 8 WMMAs per 8 ds_loads, and the two
// chains interleave to hide XDL latency. B staged block-cooperatively into
// LDS in <=32KB chunks. A-fragments use contiguous k = 16*half + i, as B does.
// ---------------------------------------------------------------------------
template <int NT, int LDO, bool RELU>
__global__ void __launch_bounds__(256)
ttm_gemm128_t(const float* __restrict__ X, const _Float16* __restrict__ Bp,
              const float* __restrict__ bias, float* __restrict__ Out, int M) {
    constexpr int CHUNK = (NT < 8) ? NT : 8;
    __shared__ _Float16 sB[CHUNK * 2048];      // CHUNK n-tiles * 4KB

    int wave = threadIdx.x >> 5;
    int lane = threadIdx.x & 31;
    int r0 = (blockIdx.x * 8 + wave) * 32;     // 2 strips per wave
    bool act0 = (r0 < M);                      // wave-uniform
    bool act1 = (r0 + 16 < M);                 // wave-uniform
    int mrow = lane & 15;
    int half = lane >> 4;
    int kb = half * 16;

    v16h a0[4] = {}, a1[4] = {};
    if (act0) {
        const float* xrow = X + (long)(r0 + mrow) * 128;
        #pragma unroll
        for (int kt = 0; kt < 4; ++kt) {
            const float* p0 = xrow + kt * 32 + kb;
            const float* p1 = xrow + kt * 32 + 8 + kb;
            #pragma unroll
            for (int i = 0; i < 8; ++i) {
                a0[kt][i]     = (_Float16)p0[i];
                a0[kt][8 + i] = (_Float16)p1[i];
            }
        }
    }
    if (act1) {
        const float* xrow = X + (long)(r0 + 16 + mrow) * 128;
        #pragma unroll
        for (int kt = 0; kt < 4; ++kt) {
            const float* p0 = xrow + kt * 32 + kb;
            const float* p1 = xrow + kt * 32 + 8 + kb;
            #pragma unroll
            for (int i = 0; i < 8; ++i) {
                a1[kt][i]     = (_Float16)p0[i];
                a1[kt][8 + i] = (_Float16)p1[i];
            }
        }
    }

    #pragma unroll 1
    for (int ntBase = 0; ntBase < NT; ntBase += CHUNK) {
        __syncthreads();                       // protect LDS reuse across chunks
        {   // cooperative stage: CHUNK * 256 uint4 (16B) elements
            const uint4* src = (const uint4*)(Bp + (size_t)ntBase * 2048);
            uint4* dst = (uint4*)sB;
            #pragma unroll
            for (int i = threadIdx.x; i < CHUNK * 256; i += 256) dst[i] = src[i];
        }
        __syncthreads();

        if (act0) {
            #pragma unroll
            for (int nt = 0; nt < CHUNK; ++nt) {
                const _Float16* bp = sB + (((nt * 4) * 32 + lane) * 16);
                v16h b0 = *(const v16h*)(bp);
                v16h b1 = *(const v16h*)(bp + 512);
                v16h b2 = *(const v16h*)(bp + 1024);
                v16h b3 = *(const v16h*)(bp + 1536);
                v8f c0 = {}, c1 = {};
                c0 = __builtin_amdgcn_wmma_f32_16x16x32_f16(false, a0[0], false, b0, (short)0, c0, false, false);
                c1 = __builtin_amdgcn_wmma_f32_16x16x32_f16(false, a1[0], false, b0, (short)0, c1, false, false);
                c0 = __builtin_amdgcn_wmma_f32_16x16x32_f16(false, a0[1], false, b1, (short)0, c0, false, false);
                c1 = __builtin_amdgcn_wmma_f32_16x16x32_f16(false, a1[1], false, b1, (short)0, c1, false, false);
                c0 = __builtin_amdgcn_wmma_f32_16x16x32_f16(false, a0[2], false, b2, (short)0, c0, false, false);
                c1 = __builtin_amdgcn_wmma_f32_16x16x32_f16(false, a1[2], false, b2, (short)0, c1, false, false);
                c0 = __builtin_amdgcn_wmma_f32_16x16x32_f16(false, a0[3], false, b3, (short)0, c0, false, false);
                c1 = __builtin_amdgcn_wmma_f32_16x16x32_f16(false, a1[3], false, b3, (short)0, c1, false, false);

                int ncol = (ntBase + nt) * 16 + mrow;
                float bv = bias[ncol];
                float* p0 = Out + (long)(r0 + 8 * half) * LDO + ncol;
                #pragma unroll
                for (int j = 0; j < 8; ++j) {
                    float v = c0[j] + bv;
                    if (RELU) v = fmaxf(v, 0.0f);
                    p0[(long)j * LDO] = v;
                }
                if (act1) {
                    float* p1 = p0 + 16 * LDO;
                    #pragma unroll
                    for (int j = 0; j < 8; ++j) {
                        float v = c1[j] + bv;
                        if (RELU) v = fmaxf(v, 0.0f);
                        p1[(long)j * LDO] = v;
                    }
                }
            }
        }
    }
}

// ---------------------------------------------------------------------------
// SpMM: Y[rows[e]] += val[e] * X[cols[e]]  (one wave / edge, float4 per lane)
// ---------------------------------------------------------------------------
__global__ void __launch_bounds__(256)
ttm_spmm(const int* __restrict__ rows, const int* __restrict__ cols,
         const float* __restrict__ val, const float* __restrict__ X,
         float* __restrict__ Y, int E) {
    int e = blockIdx.x * 8 + (threadIdx.x >> 5);
    if (e >= E) return;
    int lane = threadIdx.x & 31;
    int r = rows[e], csrc = cols[e];
    float v = val[e];
    float4 xv = ((const float4*)(X + (long)csrc * 128))[lane];
    float* yp = Y + (long)r * 128 + lane * 4;
    atomicAdd(yp + 0, v * xv.x);
    atomicAdd(yp + 1, v * xv.y);
    atomicAdd(yp + 2, v * xv.z);
    atomicAdd(yp + 3, v * xv.w);
}

// ---------------------------------------------------------------------------
// Elementwise (+bias) (pre-relu) BN (post-relu); C = 128 fixed.
// ---------------------------------------------------------------------------
__global__ void __launch_bounds__(256)
ttm_bn_act(const float* __restrict__ in, const float* __restrict__ addb,
           const float* __restrict__ g, const float* __restrict__ b,
           const float* __restrict__ mu, const float* __restrict__ var,
           float* __restrict__ out, long total, int pre_relu, int post_relu) {
    long idx = (long)blockIdx.x * blockDim.x + threadIdx.x;
    if (idx >= total) return;
    int c = (int)(idx & 127);
    float x = in[idx];
    if (addb) x += addb[c];
    if (pre_relu) x = fmaxf(x, 0.0f);
    float y = (x - mu[c]) * (g[c] * rsqrtf(var[c] + 1e-5f)) + b[c];
    if (post_relu) y = fmaxf(y, 0.0f);
    out[idx] = y;
}

// global_add_pool: out[seg[n]] += X[n]
__global__ void __launch_bounds__(256)
ttm_segpool(const float* __restrict__ X, const int* __restrict__ seg,
            float* __restrict__ out, int M) {
    int n = blockIdx.x * 8 + (threadIdx.x >> 5);
    if (n >= M) return;
    int lane = threadIdx.x & 31;
    int s = seg[n];
    float4 xv = ((const float4*)(X + (long)n * 128))[lane];
    float* op = out + (long)s * 128 + lane * 4;
    atomicAdd(op + 0, xv.x);
    atomicAdd(op + 1, xv.y);
    atomicAdd(op + 2, xv.z);
    atomicAdd(op + 3, xv.w);
}

// out[n] = dot(X[n,:128], a)
__global__ void __launch_bounds__(256)
ttm_rowdot(const float* __restrict__ X, const float* __restrict__ a,
           float* __restrict__ out, int M) {
    int n = blockIdx.x * 8 + (threadIdx.x >> 5);
    if (n >= M) return;
    int lane = threadIdx.x & 31;
    float4 xv = ((const float4*)(X + (long)n * 128))[lane];
    float4 av = ((const float4*)a)[lane];
    float s = xv.x * av.x + xv.y * av.y + xv.z * av.z + xv.w * av.w;
    for (int o = 16; o; o >>= 1) s += __shfl_xor(s, o, 32);
    if (lane == 0) out[n] = s;
}

// Monotone map: unsigned compare order == float order (deterministic max)
__device__ __forceinline__ unsigned ttm_fmap(float x) {
    unsigned u = __float_as_uint(x);
    return (u >> 31) ? ~u : (u | 0x80000000u);
}
__device__ __forceinline__ float ttm_funmap(unsigned m) {
    unsigned u = (m >> 31) ? (m & 0x7fffffffu) : ~m;
    return __uint_as_float(u);
}
__device__ __forceinline__ float ttm_leaky(float a) { return a > 0.0f ? a : 0.01f * a; }

__global__ void __launch_bounds__(256)
ttm_attmax(const float* __restrict__ an, const float* __restrict__ ad,
           const int* __restrict__ seg, unsigned* __restrict__ mx, int M) {
    int n = blockIdx.x * blockDim.x + threadIdx.x;
    if (n >= M) return;
    int s = seg[n];
    float a = ttm_leaky(an[n] + ad[s]);
    atomicMax(mx + s, ttm_fmap(a));
}

__global__ void __launch_bounds__(256)
ttm_attexp(const float* __restrict__ an, const float* __restrict__ ad,
           const int* __restrict__ seg, const unsigned* __restrict__ mx,
           float* __restrict__ en, float* __restrict__ den, int M) {
    int n = blockIdx.x * blockDim.x + threadIdx.x;
    if (n >= M) return;
    int s = seg[n];
    float a = ttm_leaky(an[n] + ad[s]);
    float e = __expf(a - ttm_funmap(mx[s]));
    en[n] = e;
    atomicAdd(den + s, e);
}

__global__ void __launch_bounds__(256)
ttm_attapply(const float* __restrict__ hs, const float* __restrict__ en,
             const float* __restrict__ den, const int* __restrict__ seg,
             float* __restrict__ g, int M) {
    int n = blockIdx.x * 8 + (threadIdx.x >> 5);
    if (n >= M) return;
    int lane = threadIdx.x & 31;
    int s = seg[n];
    float w = en[n] / den[s];
    float4 hv = ((const float4*)(hs + (long)n * 128))[lane];
    float* gp = g + (long)s * 128 + lane * 4;
    atomicAdd(gp + 0, w * hv.x);
    atomicAdd(gp + 1, w * hv.y);
    atomicAdd(gp + 2, w * hv.z);
    atomicAdd(gp + 3, w * hv.w);
}

// g = elu(g + bias)   (alpha = 1)
__global__ void __launch_bounds__(256)
ttm_bias_elu(float* __restrict__ g, const float* __restrict__ bias, long total) {
    long idx = (long)blockIdx.x * blockDim.x + threadIdx.x;
    if (idx >= total) return;
    float v = g[idx] + bias[(int)(idx & 127)];
    g[idx] = v > 0.0f ? v : (__expf(v) - 1.0f);
}

__device__ __forceinline__ float ttm_sigm(float x) { return 1.0f / (1.0f + __expf(-x)); }

// GRUCell elementwise combine (gi/gh already include biases)
__global__ void __launch_bounds__(256)
ttm_gru(const float* __restrict__ gi, const float* __restrict__ gh,
        const float* __restrict__ h, float* __restrict__ out, long total) {
    long idx = (long)blockIdx.x * blockDim.x + threadIdx.x;
    if (idx >= total) return;
    int c = (int)(idx & 127);
    long t = idx >> 7;
    const float* gip = gi + t * 384;
    const float* ghp = gh + t * 384;
    float r = ttm_sigm(gip[c] + ghp[c]);
    float z = ttm_sigm(gip[128 + c] + ghp[128 + c]);
    float nn = tanhf(gip[256 + c] + r * ghp[256 + c]);
    float hv = h[idx];
    out[idx] = (1.0f - z) * nn + z * hv;
}

// out[m] = dot(H[m,:64], w2) + b2
__global__ void __launch_bounds__(256)
ttm_finaldot(const float* __restrict__ H, const float* __restrict__ w2,
             const float* __restrict__ b2, float* __restrict__ out, int M) {
    int m = blockIdx.x * 8 + (threadIdx.x >> 5);
    if (m >= M) return;
    int lane = threadIdx.x & 31;
    float s = H[(long)m * 64 + lane] * w2[lane] + H[(long)m * 64 + 32 + lane] * w2[32 + lane];
    for (int o = 16; o; o >>= 1) s += __shfl_xor(s, o, 32);
    if (lane == 0) out[m] = s + b2[0];
}

// ---------------------------------------------------------------------------
// Host-side helpers
// ---------------------------------------------------------------------------
static inline int gemm_blocks32(int M) { return (M + 255) / 256; }  // 8 waves * 32 rows
static inline int ew_blocks(long total) { return (int)((total + 255) / 256); }

static void run_stage(const float* xin, int M, const int* seg, int NS,
                      const float* bn_g, const float* bn_b, const float* bn_m, const float* bn_v,
                      int bni, int bnm, int bno,
                      const _Float16* pWs, const _Float16* pWd,
                      const float* a_s, const float* a_d, const float* gbias,
                      const _Float16* pWih, const _Float16* pWhh,
                      const float* bih, const float* bhh, const float* zbias,
                      float* xbn, float* hs, float* mean, float* hd, float* gacc,
                      float* gi, float* gh, float* an, float* en, float* ad,
                      unsigned* mx, float* den, float* out, hipStream_t stream) {
    long Mt = (long)M * 128, St = (long)NS * 128;
    // x = BN(x)
    ttm_bn_act<<<ew_blocks(Mt), 256, 0, stream>>>(xin, nullptr,
        bn_g + bni * 128, bn_b + bni * 128, bn_m + bni * 128, bn_v + bni * 128,
        xbn, Mt, 0, 0);
    // mean = BN(relu(segment_sum(x)))
    hipMemsetAsync(mean, 0, (size_t)St * 4, stream);
    ttm_segpool<<<(M + 7) / 8, 256, 0, stream>>>(xbn, seg, mean, M);
    ttm_bn_act<<<ew_blocks(St), 256, 0, stream>>>(mean, nullptr,
        bn_g + bnm * 128, bn_b + bnm * 128, bn_m + bnm * 128, bn_v + bnm * 128,
        mean, St, 1, 0);
    // GAT transforms
    ttm_gemm128_t<8, 128, false><<<gemm_blocks32(M), 256, 0, stream>>>(xbn, pWs, zbias, hs, M);
    ttm_gemm128_t<8, 128, false><<<gemm_blocks32(NS), 256, 0, stream>>>(mean, pWd, zbias, hd, NS);
    ttm_rowdot<<<(M + 7) / 8, 256, 0, stream>>>(hs, a_s, an, M);
    ttm_rowdot<<<(NS + 7) / 8, 256, 0, stream>>>(hd, a_d, ad, NS);
    // segment softmax
    hipMemsetAsync(mx, 0, (size_t)NS * 4, stream);   // mapped-uint 0 == -inf-like
    hipMemsetAsync(den, 0, (size_t)NS * 4, stream);
    ttm_attmax<<<ew_blocks(M), 256, 0, stream>>>(an, ad, seg, mx, M);
    ttm_attexp<<<ew_blocks(M), 256, 0, stream>>>(an, ad, seg, mx, en, den, M);
    hipMemsetAsync(gacc, 0, (size_t)St * 4, stream);
    ttm_attapply<<<(M + 7) / 8, 256, 0, stream>>>(hs, en, den, seg, gacc, M);
    // g = elu(g + bias)
    ttm_bias_elu<<<ew_blocks(St), 256, 0, stream>>>(gacc, gbias, St);
    // GRU
    ttm_gemm128_t<24, 384, false><<<gemm_blocks32(NS), 256, 0, stream>>>(gacc, pWih, bih, gi, NS);
    ttm_gemm128_t<24, 384, false><<<gemm_blocks32(NS), 256, 0, stream>>>(mean, pWhh, bhh, gh, NS);
    ttm_gru<<<ew_blocks(St), 256, 0, stream>>>(gi, gh, mean, out, St);
    // out = BN(relu(out))
    ttm_bn_act<<<ew_blocks(St), 256, 0, stream>>>(out, nullptr,
        bn_g + bno * 128, bn_b + bno * 128, bn_m + bno * 128, bn_v + bno * 128,
        out, St, 1, 0);
}

extern "C" void kernel_launch(void* const* d_in, const int* in_sizes, int n_in,
                              void* d_out, int out_size, void* d_ws, size_t ws_size,
                              hipStream_t stream) {
    const float* node_attr = (const float*)d_in[0];
    const int*   adj       = (const int*)d_in[1];   // [2,E]
    const float* adj_val   = (const float*)d_in[2];
    const int*   tt_node   = (const int*)d_in[3];
    const int*   tt_graph  = (const int*)d_in[4];
    const float* W_gcn     = (const float*)d_in[5];
    const float* b_gcn     = (const float*)d_in[6];
    const float* bn_g      = (const float*)d_in[7];
    const float* bn_b      = (const float*)d_in[8];
    const float* bn_m      = (const float*)d_in[9];
    const float* bn_v      = (const float*)d_in[10];
    const float* gWs       = (const float*)d_in[11];
    const float* gWd       = (const float*)d_in[12];
    const float* gas       = (const float*)d_in[13];
    const float* gad       = (const float*)d_in[14];
    const float* gbias     = (const float*)d_in[15];
    const float* Wih       = (const float*)d_in[16];
    const float* Whh       = (const float*)d_in[17];
    const float* bih       = (const float*)d_in[18];
    const float* bhh       = (const float*)d_in[19];
    const float* pW1       = (const float*)d_in[20];
    const float* pb1       = (const float*)d_in[21];
    const float* pW2       = (const float*)d_in[22];
    const float* pb2       = (const float*)d_in[23];

    // --- workspace layout (bump allocator, 256B aligned) ---
    char* base = (char*)d_ws;
    size_t off = 0;
    auto alloc = [&](size_t bytes) -> void* {
        void* r = base + off;
        off += (bytes + 255) & ~(size_t)255;
        return r;
    };
    float* fT   = (float*)alloc((size_t)NN * 128 * 4);     // gemm temp
    float* fY   = (float*)alloc((size_t)NN * 128 * 4);     // spmm accum / hs
    float* fX   = (float*)alloc((size_t)NN * 128 * 4);     // current node feats
    float* mean = (float*)alloc((size_t)NTTs * 128 * 4);
    float* hd   = (float*)alloc((size_t)NTTs * 128 * 4);
    float* gacc = (float*)alloc((size_t)NTTs * 128 * 4);
    float* giB  = (float*)alloc((size_t)NTTs * 384 * 4);
    float* ghB  = (float*)alloc((size_t)NTTs * 384 * 4);
    float* ttout  = (float*)alloc((size_t)NTTs * 128 * 4);
    float* molout = (float*)alloc((size_t)NMOLs * 128 * 4);
    float* hpred  = (float*)alloc((size_t)NMOLs * 64 * 4);
    float* an  = (float*)alloc((size_t)NN * 4);
    float* en  = (float*)alloc((size_t)NN * 4);
    float* ad  = (float*)alloc((size_t)NTTs * 4);
    unsigned* mx = (unsigned*)alloc((size_t)NTTs * 4);
    float* den = (float*)alloc((size_t)NTTs * 4);
    float* zbias = (float*)alloc(384 * 4);                 // always-valid zero bias
    _Float16* pWg[2], * pWsP[2], * pWdP[2], * pWihP[2], * pWhhP[2];
    for (int i = 0; i < 2; ++i) {
        pWg[i]   = (_Float16*)alloc(8 * 4096);
        pWsP[i]  = (_Float16*)alloc(8 * 4096);
        pWdP[i]  = (_Float16*)alloc(8 * 4096);
        pWihP[i] = (_Float16*)alloc(24 * 4096);
        pWhhP[i] = (_Float16*)alloc(24 * 4096);
    }
    _Float16* ppW1 = (_Float16*)alloc(4 * 4096);

    hipMemsetAsync(zbias, 0, 384 * 4, stream);

    // --- pack all weights into WMMA B-fragment layout ---
    for (int i = 0; i < 2; ++i) {
        ttm_pack_w<<<8, 32, 0, stream>>>(W_gcn + (size_t)i * 128 * 128, pWg[i]);
        ttm_pack_w<<<8, 32, 0, stream>>>(gWs   + (size_t)i * 128 * 128, pWsP[i]);
        ttm_pack_w<<<8, 32, 0, stream>>>(gWd   + (size_t)i * 128 * 128, pWdP[i]);
        ttm_pack_w<<<24, 32, 0, stream>>>(Wih  + (size_t)i * 384 * 128, pWihP[i]);
        ttm_pack_w<<<24, 32, 0, stream>>>(Whh  + (size_t)i * 384 * 128, pWhhP[i]);
    }
    ttm_pack_w<<<4, 32, 0, stream>>>(pW1, ppW1);

    // --- GCN layers: x = relu(BN(A @ (x W^T) + b)) ---
    for (int i = 0; i < 2; ++i) {
        const float* src = (i == 0) ? node_attr : fX;
        ttm_gemm128_t<8, 128, false><<<gemm_blocks32(NN), 256, 0, stream>>>(src, pWg[i], zbias, fT, NN);
        hipMemsetAsync(fY, 0, (size_t)NN * 128 * 4, stream);
        ttm_spmm<<<(EE + 7) / 8, 256, 0, stream>>>(adj, adj + EE, adj_val, fT, fY, EE);
        ttm_bn_act<<<ew_blocks((long)NN * 128), 256, 0, stream>>>(fY, b_gcn + i * 128,
            bn_g + i * 128, bn_b + i * 128, bn_m + i * 128, bn_v + i * 128,
            fX, (long)NN * 128, 0, 1);
    }

    // --- tt pooling stage: nodes -> tt subgraphs ---
    run_stage(fX, NN, tt_node, NTTs,
              bn_g, bn_b, bn_m, bn_v, 2, 3, 4,
              pWsP[0], pWdP[0], gas + 0, gad + 0, gbias + 0,
              pWihP[0], pWhhP[0], bih + 0, bhh + 0, zbias,
              fT, fY, mean, hd, gacc, giB, ghB, an, en, ad, mx, den,
              ttout, stream);

    // --- mol pooling stage: tt -> molecules ---
    run_stage(ttout, NTTs, tt_graph, NMOLs,
              bn_g, bn_b, bn_m, bn_v, 5, 6, 7,
              pWsP[1], pWdP[1], gas + 128, gad + 128, gbias + 128,
              pWihP[1], pWhhP[1], bih + 384, bhh + 384, zbias,
              fT, fY, mean, hd, gacc, giB, ghB, an, en, ad, mx, den,
              molout, stream);

    // --- predictor: relu(mol @ pW1^T + pb1) @ pW2^T + pb2 ---
    ttm_gemm128_t<4, 64, true><<<gemm_blocks32(NMOLs), 256, 0, stream>>>(molout, ppW1, pb1, hpred, NMOLs);
    ttm_finaldot<<<(NMOLs + 7) / 8, 256, 0, stream>>>(hpred, pW2, pb2, (float*)d_out, NMOLs);
}
